// GNN_graphpred_31593779429494
// MI455X (gfx1250) — matrix operand order repaired and checked by reading, not compiled, not run on hardware
//
#include <hip/hip_runtime.h>
#include <stdint.h>

#define NN 200000
#define EE 600000
#define DH 128
#define DH2 256
#define LL 5
#define GG 2000
#define BN_EPS 1e-5f

typedef __attribute__((ext_vector_type(16))) __bf16 v16bf;
typedef __attribute__((ext_vector_type(8)))  float  v8f;
typedef __attribute__((ext_vector_type(4)))  float  f4;   // native vec for NT builtins

union FragAB {
    v16bf v;
    uint32_t u[8];
    uint4 q[2];
};

// native hardware bf16 convert (RNE) instead of manual integer rounding
__device__ __forceinline__ unsigned short f2bf(float x) {
    __bf16 b = (__bf16)x;
    return __builtin_bit_cast(unsigned short, b);
}

// ---------------------------------------------------------------------------
// Pack W1 (L x 128 x 256 fp32, row-major k,n) into bf16 WMMA B-fragments.
// Fragment (kt,nt): lane<16 -> col n=nt*16+lane, K=kt*32+0..15
//                   lane>=16 -> col n=nt*16+lane-16, K=kt*32+16..31
// Storage: [((l*64 + kt*16+nt)*32 + lane)*16 + i]  (i = K within half)
// ---------------------------------------------------------------------------
__global__ void pack_w1_kernel(const float* __restrict__ W1, unsigned short* __restrict__ w1f) {
    int t = blockIdx.x * blockDim.x + threadIdx.x;
    if (t >= LL * 32768) return;
    int l    = t / 32768;
    int rem  = t - l * 32768;
    int fi   = rem >> 9;            // kt*16+nt  (0..63)
    int lane = (rem >> 4) & 31;
    int i    = rem & 15;
    int kt = fi >> 4, nt = fi & 15;
    int n = nt * 16 + (lane & 15);
    int k = kt * 32 + ((lane >= 16) ? 16 : 0) + i;
    float v = W1[((size_t)l * DH + k) * DH2 + n];
    w1f[(((size_t)l * 64 + fi) * 32 + lane) * 16 + i] = f2bf(v);
}

// W2 (L x 256 x 128): kt in 0..7, nt in 0..7
__global__ void pack_w2_kernel(const float* __restrict__ W2, unsigned short* __restrict__ w2f) {
    int t = blockIdx.x * blockDim.x + threadIdx.x;
    if (t >= LL * 32768) return;
    int l    = t / 32768;
    int rem  = t - l * 32768;
    int fi   = rem >> 9;            // kt*8+nt (0..63)
    int lane = (rem >> 4) & 31;
    int i    = rem & 15;
    int kt = fi >> 3, nt = fi & 7;
    int n = nt * 16 + (lane & 15);
    int k = kt * 32 + ((lane >= 16) ? 16 : 0) + i;
    float v = W2[((size_t)l * DH2 + k) * DH + n];
    w2f[(((size_t)l * 64 + fi) * 32 + lane) * 16 + i] = f2bf(v);
}

// ---------------------------------------------------------------------------
// h = atom_emb1[x0] + atom_emb2[x1]   (thread = node x 4-dim chunk)
// ---------------------------------------------------------------------------
__global__ void embed_kernel(const int* __restrict__ x_idx,
                             const float* __restrict__ a1,
                             const float* __restrict__ a2,
                             float* __restrict__ h) {
    int t = blockIdx.x * blockDim.x + threadIdx.x;
    if (t >= NN * 32) return;
    int n  = t >> 5;
    int c4 = (t & 31) * 4;
    int x0 = x_idx[2 * n], x1 = x_idx[2 * n + 1];
    f4 u = *(const f4*)&a1[(size_t)x0 * DH + c4];
    f4 w = *(const f4*)&a2[(size_t)x1 * DH + c4];
    *(f4*)&h[(size_t)n * DH + c4] = u + w;
}

// agg = h + (edge_emb1[l][4] + edge_emb2[l][0]) broadcast
__global__ void agg_init_kernel(const float* __restrict__ h,
                                const float* __restrict__ ce1,   // edge_emb1[l][4]
                                const float* __restrict__ ce2,   // edge_emb2[l][0]
                                float* __restrict__ agg) {
    int t = blockIdx.x * blockDim.x + threadIdx.x;
    if (t >= NN * 32) return;
    int c4 = (t & 31) * 4;
    f4 a = *(const f4*)&h[(size_t)t * 4];
    f4 b = *(const f4*)&ce1[c4];
    f4 c = *(const f4*)&ce2[c4];
    *(f4*)&agg[(size_t)t * 4] = a + b + c;
}

// agg[dst] += h[src] + e1[ea0] + e2[ea1]   (thread = edge x 4-dim chunk)
__global__ void edge_scatter_kernel(const float* __restrict__ h,
                                    const int* __restrict__ edge_index,
                                    const int* __restrict__ edge_attr,
                                    const float* __restrict__ e1,   // edge_emb1[l] (6x128)
                                    const float* __restrict__ e2,   // edge_emb2[l] (3x128)
                                    float* __restrict__ agg) {
    int t = blockIdx.x * blockDim.x + threadIdx.x;
    if (t >= EE * 32) return;
    int e  = t >> 5;
    int c4 = (t & 31) * 4;
    int src = edge_index[e];
    int dst = edge_index[EE + e];
    int ea0 = edge_attr[2 * e];
    int ea1 = edge_attr[2 * e + 1];
    f4 a = *(const f4*)&h[(size_t)src * DH + c4];
    f4 b = *(const f4*)&e1[(size_t)ea0 * DH + c4];
    f4 c = *(const f4*)&e2[(size_t)ea1 * DH + c4];
    f4 s = a + b + c;
    float* p = &agg[(size_t)dst * DH + c4];
    atomicAdd(p + 0, s.x);
    atomicAdd(p + 1, s.y);
    atomicAdd(p + 2, s.z);
    atomicAdd(p + 3, s.w);
}

__global__ void zero_kernel(float* __restrict__ p, int n) {
    int t = blockIdx.x * blockDim.x + threadIdx.x;
    if (t < n) p[t] = 0.0f;
}

// ---------------------------------------------------------------------------
// Fused MLP: z = relu(agg @ W1 + b1) @ W2 + b2, plus BN statistics.
// 4 waves/block, each wave owns a 16-row tile (block = 64 rows).
// agg read and zbuf write are single-use streams -> non-temporal hints.
// ---------------------------------------------------------------------------
__global__ __launch_bounds__(128) void mlp_kernel(
    const float* __restrict__ agg,
    const unsigned short* __restrict__ w1f,   // layer slice: 64 frags
    const unsigned short* __restrict__ w2f,   // layer slice: 64 frags
    const float* __restrict__ b1,             // 256
    const float* __restrict__ b2,             // 128
    float* __restrict__ zbuf,
    float* __restrict__ stats)                // sum[128], sumsq[128]
{
    __shared__ alignas(16) unsigned short aStage[4][16][DH];    // 16 KB
    __shared__ alignas(16) unsigned short z1Stage[4][16][DH2];  // 32 KB (reused as f32 tile)
    __shared__ float sSum[DH];
    __shared__ float sSq[DH];

    const int lane = threadIdx.x & 31;
    const int wave = threadIdx.x >> 5;
    if (threadIdx.x < DH) { sSum[threadIdx.x] = 0.0f; sSq[threadIdx.x] = 0.0f; }
    __syncthreads();

    const int row0 = blockIdx.x * 64 + wave * 16;

    // Stage A tile: 16 rows x 128 fp32 -> bf16 LDS (coalesced float4 loads, NT)
    #pragma unroll
    for (int i = 0; i < 16; ++i) {
        int idx = lane + 32 * i;            // 512 float4 per wave tile
        int r   = idx >> 5;
        int c4  = (idx & 31) * 4;
        f4 f = __builtin_nontemporal_load(
            (const f4*)&agg[(size_t)(row0 + r) * DH + c4]);
        aStage[wave][r][c4 + 0] = f2bf(f.x);
        aStage[wave][r][c4 + 1] = f2bf(f.y);
        aStage[wave][r][c4 + 2] = f2bf(f.z);
        aStage[wave][r][c4 + 3] = f2bf(f.w);
    }

    const int m  = lane & 15;
    const int kb = (lane >> 4) << 4;        // 0 or 16 (A-layout half-wave K base)
    const int col16 = lane & 15;
    const int rbase = (lane >> 4) * 8;      // C-layout row base

    // Build 4 A fragments (K = 0..127)
    FragAB a1[4];
    #pragma unroll
    for (int kt = 0; kt < 4; ++kt) {
        #pragma unroll
        for (int v = 0; v < 8; ++v) {
            int k = kt * 32 + kb + 2 * v;
            a1[kt].u[v] = *(const uint32_t*)&aStage[wave][m][k];
        }
    }

    // GEMM1: 16 output tiles of 16 cols, relu -> bf16 into z1Stage
    for (int nt = 0; nt < 16; ++nt) {
        v8f acc = {};
        #pragma unroll
        for (int kt = 0; kt < 4; ++kt) {
            FragAB b;
            const uint4* p = (const uint4*)(w1f + ((size_t)(kt * 16 + nt) * 32 + lane) * 16);
            b.q[0] = p[0]; b.q[1] = p[1];
            acc = __builtin_amdgcn_wmma_f32_16x16x32_bf16(
                false, a1[kt].v, false, b.v, (short)0, acc, false, false);
        }
        float bb = b1[nt * 16 + col16];
        int colBase = nt * 16 + col16;
        #pragma unroll
        for (int r = 0; r < 8; ++r) {
            float zv = fmaxf(acc[r] + bb, 0.0f);
            z1Stage[wave][rbase + r][colBase] = f2bf(zv);
        }
    }

    // Build 8 A fragments for GEMM2 (K = 0..255) from z1Stage
    FragAB a2[8];
    #pragma unroll
    for (int kt = 0; kt < 8; ++kt) {
        #pragma unroll
        for (int v = 0; v < 8; ++v) {
            int k = kt * 32 + kb + 2 * v;
            a2[kt].u[v] = *(const uint32_t*)&z1Stage[wave][m][k];
        }
    }

    // Reuse z1Stage region as a 16x128 fp32 staging tile (exactly 8 KB/wave)
    float* ztile = (float*)&z1Stage[wave][0][0];

    // GEMM2: 8 output tiles; BN partial stats per column
    for (int nt = 0; nt < 8; ++nt) {
        v8f acc = {};
        #pragma unroll
        for (int kt = 0; kt < 8; ++kt) {
            FragAB b;
            const uint4* p = (const uint4*)(w2f + ((size_t)(kt * 8 + nt) * 32 + lane) * 16);
            b.q[0] = p[0]; b.q[1] = p[1];
            acc = __builtin_amdgcn_wmma_f32_16x16x32_bf16(
                false, a2[kt].v, false, b.v, (short)0, acc, false, false);
        }
        float bb = b2[nt * 16 + col16];
        int col = nt * 16 + col16;
        float ls = 0.0f, lq = 0.0f;
        #pragma unroll
        for (int r = 0; r < 8; ++r) {
            float zv = acc[r] + bb;
            ztile[(rbase + r) * DH + col] = zv;
            ls += zv;
            lq += zv * zv;
        }
        atomicAdd(&sSum[col], ls);
        atomicAdd(&sSq[col], lq);
    }

    // Coalesced z tile write-back (NT stream)
    #pragma unroll
    for (int i = 0; i < 16; ++i) {
        int idx = lane + 32 * i;
        int r   = idx >> 5;
        int c4  = (idx & 31) * 4;
        f4 v = *(const f4*)&ztile[r * DH + c4];
        __builtin_nontemporal_store(v, (f4*)&zbuf[(size_t)(row0 + r) * DH + c4]);
    }

    __syncthreads();
    if (threadIdx.x < DH) {
        atomicAdd(&stats[threadIdx.x],      sSum[threadIdx.x]);
        atomicAdd(&stats[DH + threadIdx.x], sSq[threadIdx.x]);
    }
}

// scale/shift for BN
__global__ void bn_finalize_kernel(const float* __restrict__ stats,
                                   const float* __restrict__ gamma,
                                   const float* __restrict__ beta,
                                   float* __restrict__ bnp) {
    int c = threadIdx.x;
    if (c >= DH) return;
    const float invN = 1.0f / (float)NN;
    float mu  = stats[c] * invN;
    float var = stats[DH + c] * invN - mu * mu;
    float sc  = rsqrtf(var + BN_EPS) * gamma[c];
    bnp[c]      = sc;
    bnp[DH + c] = beta[c] - mu * sc;
}

__global__ void bn_apply_kernel(const float* __restrict__ zbuf,
                                const float* __restrict__ bnp,
                                float* __restrict__ h, int do_relu) {
    int t = blockIdx.x * blockDim.x + threadIdx.x;
    if (t >= NN * 32) return;
    int c4 = (t & 31) * 4;
    f4 z = __builtin_nontemporal_load((const f4*)&zbuf[(size_t)t * 4]);
    f4 sc = *(const f4*)&bnp[c4];
    f4 sh = *(const f4*)&bnp[DH + c4];
    f4 r = z * sc + sh;
    if (do_relu) {
        r.x = fmaxf(r.x, 0.0f);
        r.y = fmaxf(r.y, 0.0f);
        r.z = fmaxf(r.z, 0.0f);
        r.w = fmaxf(r.w, 0.0f);
    }
    *(f4*)&h[(size_t)t * 4] = r;
}

__global__ void pool_scatter_kernel(const float* __restrict__ h,
                                    const int* __restrict__ batch,
                                    float* __restrict__ pooled) {
    int t = blockIdx.x * blockDim.x + threadIdx.x;
    if (t >= NN * 32) return;
    int n  = t >> 5;
    int c4 = (t & 31) * 4;
    int g  = batch[n];
    f4 v = *(const f4*)&h[(size_t)n * DH + c4];
    float* p = &pooled[(size_t)g * DH + c4];
    atomicAdd(p + 0, v.x);
    atomicAdd(p + 1, v.y);
    atomicAdd(p + 2, v.z);
    atomicAdd(p + 3, v.w);
}

__global__ void predict_kernel(const float* __restrict__ pooled,
                               const float* __restrict__ Wp,
                               const float* __restrict__ bp,
                               float* __restrict__ out) {
    __shared__ float red[DH];
    int g = blockIdx.x;
    int t = threadIdx.x;
    red[t] = pooled[(size_t)g * DH + t] * Wp[t];
    __syncthreads();
    for (int s = 64; s > 0; s >>= 1) {
        if (t < s) red[t] += red[t + s];
        __syncthreads();
    }
    if (t == 0) out[g] = red[0] + bp[0];
}

// ---------------------------------------------------------------------------
extern "C" void kernel_launch(void* const* d_in, const int* in_sizes, int n_in,
                              void* d_out, int out_size, void* d_ws, size_t ws_size,
                              hipStream_t stream) {
    const int*   x_idx      = (const int*)d_in[0];
    const int*   edge_index = (const int*)d_in[1];
    const int*   edge_attr  = (const int*)d_in[2];
    const int*   batch      = (const int*)d_in[3];
    const float* atom1      = (const float*)d_in[4];
    const float* atom2      = (const float*)d_in[5];
    const float* ee1        = (const float*)d_in[6];   // (L,6,128)
    const float* ee2        = (const float*)d_in[7];   // (L,3,128)
    const float* W1         = (const float*)d_in[8];   // (L,128,256)
    const float* b1         = (const float*)d_in[9];   // (L,256)
    const float* W2         = (const float*)d_in[10];  // (L,256,128)
    const float* b2         = (const float*)d_in[11];  // (L,128)
    const float* gamma      = (const float*)d_in[12];
    const float* beta       = (const float*)d_in[13];
    const float* Wp         = (const float*)d_in[14];  // (128,1)
    const float* bp         = (const float*)d_in[15];  // (1,)
    float* out = (float*)d_out;

    // workspace layout (256-byte aligned slabs)
    char* ws = (char*)d_ws;
    size_t off = 0;
    auto alloc = [&](size_t bytes) { void* p = ws + off; off += (bytes + 255) & ~(size_t)255; return p; };
    float* h      = (float*)alloc((size_t)NN * DH * 4);
    float* agg    = (float*)alloc((size_t)NN * DH * 4);
    float* zbuf   = (float*)alloc((size_t)NN * DH * 4);
    unsigned short* w1f = (unsigned short*)alloc((size_t)LL * 64 * 32 * 16 * 2);
    unsigned short* w2f = (unsigned short*)alloc((size_t)LL * 64 * 32 * 16 * 2);
    float* stats  = (float*)alloc(2 * DH * 4);
    float* bnp    = (float*)alloc(2 * DH * 4);
    float* pooled = (float*)alloc((size_t)GG * DH * 4);
    (void)ws_size; (void)n_in; (void)in_sizes; (void)out_size;

    // pack weights into WMMA B-fragment layout (bf16)
    pack_w1_kernel<<<(LL * 32768 + 255) / 256, 256, 0, stream>>>(W1, w1f);
    pack_w2_kernel<<<(LL * 32768 + 255) / 256, 256, 0, stream>>>(W2, w2f);

    // h0
    embed_kernel<<<(NN * 32 + 255) / 256, 256, 0, stream>>>(x_idx, atom1, atom2, h);

    for (int l = 0; l < LL; ++l) {
        zero_kernel<<<1, 256, 0, stream>>>(stats, 2 * DH);
        agg_init_kernel<<<(NN * 32 + 255) / 256, 256, 0, stream>>>(
            h, ee1 + ((size_t)l * 6 + 4) * DH, ee2 + (size_t)l * 3 * DH, agg);
        edge_scatter_kernel<<<(EE * 32 + 255) / 256, 256, 0, stream>>>(
            h, edge_index, edge_attr, ee1 + (size_t)l * 6 * DH, ee2 + (size_t)l * 3 * DH, agg);
        mlp_kernel<<<NN / 64, 128, 0, stream>>>(
            agg, w1f + (size_t)l * 32768, w2f + (size_t)l * 32768,
            b1 + (size_t)l * DH2, b2 + (size_t)l * DH, zbuf, stats);
        bn_finalize_kernel<<<1, 128, 0, stream>>>(stats, gamma + (size_t)l * DH, beta + (size_t)l * DH, bnp);
        bn_apply_kernel<<<(NN * 32 + 255) / 256, 256, 0, stream>>>(zbuf, bnp, h, (l < LL - 1) ? 1 : 0);
    }

    zero_kernel<<<(GG * DH + 255) / 256, 256, 0, stream>>>(pooled, GG * DH);
    pool_scatter_kernel<<<(NN * 32 + 255) / 256, 256, 0, stream>>>(h, batch, pooled);
    predict_kernel<<<GG, 128, 0, stream>>>(pooled, Wp, bp, out);
}
